// MultiheadAttention_44375602102607
// MI455X (gfx1250) — compile-verified, matrix operand order checked by reading, not run on hardware
//
#include <hip/hip_runtime.h>
#include <hip/hip_bf16.h>
#include <math.h>

#define L_SEQ 2048
#define DIMM  1024
#define NHEAD 16
#define HDIM  64

typedef __attribute__((ext_vector_type(16))) __bf16       v16bf;
typedef __attribute__((ext_vector_type(8)))  float        v8f;
typedef __attribute__((ext_vector_type(4)))  unsigned int v4u;

union FragU { v16bf v; v4u q[2]; unsigned short us[16]; };

__device__ __forceinline__ unsigned short f2b(float f) {
  union { float f; unsigned u; } x; x.f = f;
  unsigned r = x.u + 0x7FFFu + ((x.u >> 16) & 1u);
  return (unsigned short)(r >> 16);
}
__device__ __forceinline__ float b2f(unsigned short s) {
  union { unsigned u; float f; } x; x.u = ((unsigned)s) << 16;
  return x.f;
}

#define WMMA_BF16(A, B, C) \
  __builtin_amdgcn_wmma_f32_16x16x32_bf16(false, (A), false, (B), (short)0, (C), false, false)

// 16-bit A/B fragment, row-major source: lane holds row (lane&15); half-wave
// (lane>>4) selects K-subblocks {0..7,16..23} vs {8..15,24..31} (ISA 7.12.2).
__device__ __forceinline__ v16bf frag_rm(const unsigned short* base, int stride, int lane) {
  const int r = lane & 15, hf = (lane >> 4) & 1;
  const unsigned short* p = base + r * stride + 8 * hf;
  FragU f;
  f.q[0] = *(const v4u*)(p);
  f.q[1] = *(const v4u*)(p + 16);
  return f.v;
}

// B fragment where the K index walks down rows of a row-major matrix
// (i.e. source is K-major relative to the fragment): strided ushort gathers.
__device__ __forceinline__ v16bf frag_ck(const unsigned short* base, int kstride, int lane) {
  const int n = lane & 15, hf = (lane >> 4) & 1;
  FragU f;
#pragma unroll
  for (int j = 0; j < 8; ++j) f.us[j] = base[n + (8 * hf + j) * kstride];
#pragma unroll
  for (int j = 0; j < 8; ++j) f.us[8 + j] = base[n + (16 + 8 * hf + j) * kstride];
  return f.v;
}

__global__ void k_cvt_bf16(const float* __restrict__ src, unsigned short* __restrict__ dst, int n) {
  int i = blockIdx.x * blockDim.x + threadIdx.x;
  int st = gridDim.x * blockDim.x;
  for (; i < n; i += st) dst[i] = f2b(src[i]);
}

__global__ void k_zero_f32(float* __restrict__ p, int n) {
  int i = blockIdx.x * blockDim.x + threadIdx.x;
  int st = gridDim.x * blockDim.x;
  for (; i < n; i += st) p[i] = 0.0f;
}

__global__ void k_rope(const unsigned short* __restrict__ x, unsigned short* __restrict__ y, int n) {
  int i = blockIdx.x * blockDim.x + threadIdx.x;
  int st = gridDim.x * blockDim.x;
  for (; i < n; i += st) {
    int l  = i >> 10;            // / DIMM
    int d  = i & (DIMM - 1);
    int hd = d & (HDIM - 1);
    int fi = hd & 31;
    float inv = __expf(-(2.0f * (float)fi / 64.0f) * __logf(10000.0f));
    float ang = (float)l * inv;
    float sn, cs;
    __sincosf(ang, &sn, &cs);
    float xv = b2f(x[i]);
    float xr = (hd < 32) ? -b2f(x[i + 32]) : b2f(x[i - 32]);
    y[i] = f2b(xv * cs + xr * sn);
  }
}

// C[l, n] = (sum_k A[l,k] * W[n,k] + bias[n]) * scale.
// 16x64 strip per wave, software-pipelined: fragments for k+32 are issued
// before the WMMAs for k, so WMMA waits are on already-retired loads.
__global__ void __launch_bounds__(256) k_gemm(const unsigned short* __restrict__ A,
                                              const unsigned short* __restrict__ W,
                                              const float* __restrict__ bias,
                                              int N, int K, float scale,
                                              float* __restrict__ outF,
                                              unsigned short* __restrict__ outB) {
  const int l0 = blockIdx.x * 16;
  const int wave = threadIdx.x >> 5, lane = threadIdx.x & 31;
  const int n0 = blockIdx.y * 512 + wave * 64;
  const unsigned short* Ab = A + (size_t)l0 * K;
  const unsigned short* Wb = W + (size_t)n0 * K;
  v8f vz = {};
  v8f acc[4];
#pragma unroll
  for (int nt = 0; nt < 4; ++nt) acc[nt] = vz;

  // prologue: fragments for k0 = 0
  v16bf a = frag_rm(Ab, K, lane);
  v16bf b[4];
#pragma unroll
  for (int nt = 0; nt < 4; ++nt) b[nt] = frag_rm(Wb + (size_t)(16 * nt) * K, K, lane);

  for (int k0 = 32; k0 < K; k0 += 32) {
    // issue next-stage loads first
    v16bf an = frag_rm(Ab + k0, K, lane);
    v16bf bn[4];
#pragma unroll
    for (int nt = 0; nt < 4; ++nt)
      bn[nt] = frag_rm(Wb + (size_t)(16 * nt) * K + k0, K, lane);
    __builtin_prefetch(Ab + k0 + 32, 0, 1);
#pragma unroll
    for (int nt = 0; nt < 4; ++nt)
      __builtin_prefetch(Wb + (size_t)(16 * nt) * K + k0 + 32, 0, 1);
    // compute current stage
#pragma unroll
    for (int nt = 0; nt < 4; ++nt) acc[nt] = WMMA_BF16(a, b[nt], acc[nt]);
    a = an;
#pragma unroll
    for (int nt = 0; nt < 4; ++nt) b[nt] = bn[nt];
  }
#pragma unroll
  for (int nt = 0; nt < 4; ++nt) acc[nt] = WMMA_BF16(a, b[nt], acc[nt]);

  const int n16 = lane & 15, hf = lane >> 4;
#pragma unroll
  for (int nt = 0; nt < 4; ++nt) {
    const int col = n0 + nt * 16 + n16;
    const float bc = bias ? bias[col] : 0.0f;
#pragma unroll
    for (int r = 0; r < 8; ++r) {
      int row = l0 + r + 8 * hf;
      float vv = (acc[nt][r] + bc) * scale;
      if (outF) outF[(size_t)row * N + col] = vv;
      if (outB) outB[(size_t)row * N + col] = f2b(vv);
    }
  }
}

// avg[l,m] = (Qr.Kr^T - Q.K^T)[l,m] / NHEAD over full D (== mean_h(rpe-raw)).
// 16x32 per wave (2 N-tiles), double-buffered: 6 fragments x 2 stages + 4
// accumulators stays well under the VGPR budget.
__global__ void __launch_bounds__(256) k_avg(const unsigned short* __restrict__ Q,
                                             const unsigned short* __restrict__ Kb,
                                             const unsigned short* __restrict__ Qr,
                                             const unsigned short* __restrict__ Kr,
                                             float* __restrict__ avg) {
  const int l0 = blockIdx.x * 16;
  const int wave = threadIdx.x >> 5, lane = threadIdx.x & 31;
  const int m0 = blockIdx.y * 256 + wave * 32;
  v8f vz = {};
  v8f ar[2], aq[2];
#pragma unroll
  for (int nt = 0; nt < 2; ++nt) { ar[nt] = vz; aq[nt] = vz; }

  const unsigned short* Qrb = Qr + (size_t)l0 * DIMM;
  const unsigned short* Qb_ = Q + (size_t)l0 * DIMM;

  v16bf aR = frag_rm(Qrb, DIMM, lane);
  v16bf aQ = frag_rm(Qb_, DIMM, lane);
  v16bf bR[2], bQ[2];
#pragma unroll
  for (int nt = 0; nt < 2; ++nt) {
    const size_t mrow = (size_t)(m0 + 16 * nt) * DIMM;
    bR[nt] = frag_rm(Kr + mrow, DIMM, lane);
    bQ[nt] = frag_rm(Kb + mrow, DIMM, lane);
  }

  for (int k0 = 32; k0 < DIMM; k0 += 32) {
    v16bf aRn = frag_rm(Qrb + k0, DIMM, lane);
    v16bf aQn = frag_rm(Qb_ + k0, DIMM, lane);
    v16bf bRn[2], bQn[2];
#pragma unroll
    for (int nt = 0; nt < 2; ++nt) {
      const size_t mrow = (size_t)(m0 + 16 * nt) * DIMM;
      bRn[nt] = frag_rm(Kr + mrow + k0, DIMM, lane);
      bQn[nt] = frag_rm(Kb + mrow + k0, DIMM, lane);
    }
    __builtin_prefetch(Qrb + k0 + 32, 0, 1);
    __builtin_prefetch(Qb_ + k0 + 32, 0, 1);
#pragma unroll
    for (int nt = 0; nt < 2; ++nt) {
      ar[nt] = WMMA_BF16(aR, bR[nt], ar[nt]);
      aq[nt] = WMMA_BF16(aQ, bQ[nt], aq[nt]);
    }
    aR = aRn; aQ = aQn;
#pragma unroll
    for (int nt = 0; nt < 2; ++nt) { bR[nt] = bRn[nt]; bQ[nt] = bQn[nt]; }
  }
#pragma unroll
  for (int nt = 0; nt < 2; ++nt) {
    ar[nt] = WMMA_BF16(aR, bR[nt], ar[nt]);
    aq[nt] = WMMA_BF16(aQ, bQ[nt], aq[nt]);
  }

  const int n16 = lane & 15, hf = lane >> 4;
#pragma unroll
  for (int nt = 0; nt < 2; ++nt)
#pragma unroll
    for (int r = 0; r < 8; ++r)
      avg[(size_t)(l0 + r + 8 * hf) * L_SEQ + m0 + 16 * nt + n16] =
          (ar[nt][r] - aq[nt][r]) * (1.0f / NHEAD);
}

// One block = 16 query rows, loops all heads. Two-pass online softmax,
// WMMA for scores and probs.V, LDS for stat merge / probs re-layout / O-reduce.
__global__ void __launch_bounds__(256) k_attn(const unsigned short* __restrict__ Q,
                                              const unsigned short* __restrict__ Kb,
                                              const unsigned short* __restrict__ V,
                                              const unsigned short* __restrict__ Qr,
                                              const unsigned short* __restrict__ Kr,
                                              const float* __restrict__ avg,
                                              const int* __restrict__ asym,
                                              float* __restrict__ pavg,
                                              unsigned short* __restrict__ attn) {
  const int l0 = blockIdx.x * 16;
  const int wave = threadIdx.x >> 5, lane = threadIdx.x & 31;
  const int n16 = lane & 15, hf = lane >> 4;

  __shared__ float s_stat[2][8][16];
  __shared__ float s_att[16][64];
  __shared__ __align__(16) unsigned short s_prob[8][16][32];

  int rowA[8], asymRow[8];
#pragma unroll
  for (int r = 0; r < 8; ++r) {
    rowA[r] = l0 + r + 8 * hf;
    asymRow[r] = asym[rowA[r]];
  }

  for (int h = 0; h < NHEAD; ++h) {
    const int ho = h * HDIM;
    v16bf qa0 = frag_rm(Q + (size_t)l0 * DIMM + ho, DIMM, lane);
    v16bf qa1 = frag_rm(Q + (size_t)l0 * DIMM + ho + 32, DIMM, lane);
    v16bf ra0 = frag_rm(Qr + (size_t)l0 * DIMM + ho, DIMM, lane);
    v16bf ra1 = frag_rm(Qr + (size_t)l0 * DIMM + ho + 32, DIMM, lane);

    // combined score tile (16 rows x 16 cols at mb): raw, rpe, mask, avg
    auto score_tile = [&](int mb, float sc[8]) {
      v16bf kb0 = frag_rm(Kb + (size_t)mb * DIMM + ho, DIMM, lane);
      v16bf kb1 = frag_rm(Kb + (size_t)mb * DIMM + ho + 32, DIMM, lane);
      v16bf kr0 = frag_rm(Kr + (size_t)mb * DIMM + ho, DIMM, lane);
      v16bf kr1 = frag_rm(Kr + (size_t)mb * DIMM + ho + 32, DIMM, lane);
      v8f raw = {}, rpe = {};
      raw = WMMA_BF16(qa0, kb0, raw);
      raw = WMMA_BF16(qa1, kb1, raw);
      rpe = WMMA_BF16(ra0, kr0, rpe);
      rpe = WMMA_BF16(ra1, kr1, rpe);
      const int m = mb + n16;
      const int am = asym[m];
#pragma unroll
      for (int r = 0; r < 8; ++r)
        sc[r] = (am == asymRow[r]) ? rpe[r] : (raw[r] + avg[(size_t)rowA[r] * L_SEQ + m]);
    };

    // ---- pass 1: online row max / sum (wave handles m-chunks wave+8c, 32 wide)
    float rmax[8], rsum[8];
#pragma unroll
    for (int r = 0; r < 8; ++r) { rmax[r] = -1e30f; rsum[r] = 0.0f; }

    for (int c = 0; c < 8; ++c) {
      const int m0 = (wave + 8 * c) * 32;
      float sc0[8], sc1[8];
      score_tile(m0, sc0);
      score_tile(m0 + 16, sc1);
#pragma unroll
      for (int r = 0; r < 8; ++r) {
        float tm = fmaxf(sc0[r], sc1[r]);
#pragma unroll
        for (int off = 1; off < 16; off <<= 1) tm = fmaxf(tm, __shfl_xor(tm, off, 32));
        float ts = __expf(sc0[r] - tm) + __expf(sc1[r] - tm);
#pragma unroll
        for (int off = 1; off < 16; off <<= 1) ts += __shfl_xor(ts, off, 32);
        float nm = fmaxf(rmax[r], tm);
        rsum[r] = rsum[r] * __expf(rmax[r] - nm) + ts * __expf(tm - nm);
        rmax[r] = nm;
      }
    }

    // ---- merge stats across the 8 waves
    if (n16 == 0) {
#pragma unroll
      for (int r = 0; r < 8; ++r) {
        s_stat[0][wave][r + 8 * hf] = rmax[r];
        s_stat[1][wave][r + 8 * hf] = rsum[r];
      }
    }
    __syncthreads();
    float M[8], IS[8];
#pragma unroll
    for (int r = 0; r < 8; ++r) {
      const int rr = r + 8 * hf;
      float mm = -1e30f;
#pragma unroll
      for (int w = 0; w < 8; ++w) mm = fmaxf(mm, s_stat[0][w][rr]);
      float ss = 0.0f;
#pragma unroll
      for (int w = 0; w < 8; ++w) ss += s_stat[1][w][rr] * __expf(s_stat[0][w][rr] - mm);
      M[r] = mm;
      IS[r] = 1.0f / ss;
    }
    for (int i = threadIdx.x; i < 16 * 64; i += 256) ((float*)s_att)[i] = 0.0f;
    __syncthreads();

    // ---- pass 2: probs, head-mean accumulation, probs.V
    v8f vz = {};
    v8f oacc[4];
#pragma unroll
    for (int nt = 0; nt < 4; ++nt) oacc[nt] = vz;

    for (int c = 0; c < 8; ++c) {
      const int m0 = (wave + 8 * c) * 32;
#pragma unroll
      for (int t = 0; t < 2; ++t) {
        const int mb = m0 + 16 * t;
        float sc[8];
        score_tile(mb, sc);
        const int m = mb + n16;
#pragma unroll
        for (int r = 0; r < 8; ++r) {
          float p = __expf(sc[r] - M[r]) * IS[r];
          pavg[(size_t)rowA[r] * L_SEQ + m] += p * (1.0f / NHEAD);  // race-free: 1 owner
          s_prob[wave][r + 8 * hf][16 * t + n16] = f2b(p);
        }
      }
      __syncthreads();
      v16bf pa = frag_rm(&s_prob[wave][0][0], 32, lane);
#pragma unroll
      for (int nt = 0; nt < 4; ++nt) {
        v16bf vb = frag_ck(V + (size_t)m0 * DIMM + ho + nt * 16, DIMM, lane);
        oacc[nt] = WMMA_BF16(pa, vb, oacc[nt]);
      }
      __syncthreads();
    }

    // ---- reduce O across waves, emit bf16 context tile
#pragma unroll
    for (int nt = 0; nt < 4; ++nt)
#pragma unroll
      for (int r = 0; r < 8; ++r)
        atomicAdd(&s_att[r + 8 * hf][nt * 16 + n16], oacc[nt][r]);
    __syncthreads();
    for (int i = threadIdx.x; i < 16 * 64; i += 256) {
      int rr = i >> 6, cc = i & 63;
      attn[(size_t)(l0 + rr) * DIMM + ho + cc] = f2b(s_att[rr][cc]);
    }
    __syncthreads();
  }
}

extern "C" void kernel_launch(void* const* d_in, const int* in_sizes, int n_in,
                              void* d_out, int out_size, void* d_ws, size_t ws_size,
                              hipStream_t stream) {
  (void)in_sizes; (void)n_in; (void)out_size; (void)ws_size;
  const float* query = (const float*)d_in[0];
  const int*   asym  = (const int*)d_in[1];
  const float* wq = (const float*)d_in[2];
  const float* bq = (const float*)d_in[3];
  const float* wk = (const float*)d_in[4];
  const float* bk = (const float*)d_in[5];
  const float* wv = (const float*)d_in[6];
  const float* bv = (const float*)d_in[7];
  const float* wo = (const float*)d_in[8];
  const float* bo = (const float*)d_in[9];

  char* ws = (char*)d_ws;
  const size_t MB = 1ull << 20;
  unsigned short* Xbf = (unsigned short*)(ws + 0 * MB);   // query bf16   4MB
  unsigned short* Wqb = (unsigned short*)(ws + 4 * MB);   // wq bf16      2MB
  unsigned short* Wkb = (unsigned short*)(ws + 6 * MB);
  unsigned short* Wvb = (unsigned short*)(ws + 8 * MB);
  unsigned short* Wob = (unsigned short*)(ws + 10 * MB);
  unsigned short* Qb  = (unsigned short*)(ws + 12 * MB);  // scaled q bf16 4MB
  unsigned short* Kbf = (unsigned short*)(ws + 16 * MB);
  unsigned short* Vb  = (unsigned short*)(ws + 20 * MB);
  unsigned short* Qr  = (unsigned short*)(ws + 24 * MB);  // rope(q)
  unsigned short* Kr  = (unsigned short*)(ws + 28 * MB);  // rope(k)
  float*          Avg = (float*)(ws + 32 * MB);           // L*L f32     16MB
  unsigned short* Att = (unsigned short*)(ws + 48 * MB);  // context bf16 4MB

  float* outO = (float*)d_out;                     // (L, 1, D) f32
  float* outP = outO + (size_t)L_SEQ * DIMM;       // (1, L, L) f32

  const int TB = 256;
  const int nLD = L_SEQ * DIMM;
  const int nDD = DIMM * DIMM;

  k_cvt_bf16<<<(nLD + TB - 1) / TB, TB, 0, stream>>>(query, Xbf, nLD);
  k_cvt_bf16<<<(nDD + TB - 1) / TB, TB, 0, stream>>>(wq, Wqb, nDD);
  k_cvt_bf16<<<(nDD + TB - 1) / TB, TB, 0, stream>>>(wk, Wkb, nDD);
  k_cvt_bf16<<<(nDD + TB - 1) / TB, TB, 0, stream>>>(wv, Wvb, nDD);
  k_cvt_bf16<<<(nDD + TB - 1) / TB, TB, 0, stream>>>(wo, Wob, nDD);

  dim3 blk(256);
  dim3 gP(L_SEQ / 16, DIMM / 512);
  k_gemm<<<gP, blk, 0, stream>>>(Xbf, Wqb, bq, DIMM, DIMM, 0.125f, nullptr, Qb);  // 1/sqrt(64)
  k_gemm<<<gP, blk, 0, stream>>>(Xbf, Wkb, bk, DIMM, DIMM, 1.0f, nullptr, Kbf);
  k_gemm<<<gP, blk, 0, stream>>>(Xbf, Wvb, bv, DIMM, DIMM, 1.0f, nullptr, Vb);

  k_rope<<<(nLD + TB - 1) / TB, TB, 0, stream>>>(Qb, Qr, nLD);
  k_rope<<<(nLD + TB - 1) / TB, TB, 0, stream>>>(Kbf, Kr, nLD);

  dim3 gA(L_SEQ / 16, L_SEQ / 256);
  k_avg<<<gA, blk, 0, stream>>>(Qb, Kbf, Qr, Kr, Avg);

  const int nLL = L_SEQ * L_SEQ;
  k_zero_f32<<<(nLL + TB - 1) / TB, TB, 0, stream>>>(outP, nLL);

  k_attn<<<dim3(L_SEQ / 16), blk, 0, stream>>>(Qb, Kbf, Vb, Qr, Kr, Avg, asym, outP, Att);

  k_gemm<<<gP, blk, 0, stream>>>(Att, Wob, bo, DIMM, DIMM, 1.0f, outO, nullptr);
}
